// GAT_45887430591137
// MI455X (gfx1250) — compile-verified, matrix operand order checked by reading, not run on hardware
//
#include <hip/hip_runtime.h>

// ---------------- CDNA5 WMMA types ----------------
typedef __attribute__((ext_vector_type(16))) __bf16 v16bf;
typedef __attribute__((ext_vector_type(8)))  __bf16 v8bf;
typedef __attribute__((ext_vector_type(8)))  float  v8f;

#define NNODES    20000
#define NEG_SLOPE 0.01f

__device__ __forceinline__ float lrelu(float x) { return x >= 0.0f ? x : NEG_SLOPE * x; }

// order-preserving float <-> uint map so segment-max can use integer atomicMax.
// init value 0u is below encode(-inf)=0x007fffff, so any real edge value wins.
__device__ __forceinline__ unsigned f2ord(float f) {
    unsigned b = __float_as_uint(f);
    return (b & 0x80000000u) ? ~b : (b | 0x80000000u);
}
__device__ __forceinline__ float ord2f(unsigned u) {
    return (u & 0x80000000u) ? __uint_as_float(u ^ 0x80000000u) : __uint_as_float(~u);
}

// ---------------- prep: f32 -> bf16 ----------------
__global__ void k_cvt_bf16(const float* __restrict__ in, __bf16* __restrict__ out, int n) {
    int i = blockIdx.x * blockDim.x + threadIdx.x;
    if (i < n) out[i] = (__bf16)in[i];
}

// W1 (8,256,128) f32 -> Wt1[(head*128+d)*256 + k] bf16   (row-major [N=1024][K=256])
__global__ void k_prep_w1(const float* __restrict__ W1, __bf16* __restrict__ Wt) {
    int i = blockIdx.x * blockDim.x + threadIdx.x;
    if (i >= 1024 * 256) return;
    int k = i & 255, n = i >> 8;
    int head = n >> 7, d = n & 127;
    Wt[i] = (__bf16)W1[head * (256 * 128) + k * 128 + d];
}

// W2 (1024,256) f32 -> Wt2[n*1024 + k] bf16   (row-major [N=256][K=1024])
__global__ void k_prep_w2(const float* __restrict__ W2, __bf16* __restrict__ Wt) {
    int i = blockIdx.x * blockDim.x + threadIdx.x;
    if (i >= 256 * 1024) return;
    int k = i & 1023, n = i >> 10;
    Wt[i] = (__bf16)W2[k * 256 + n];
}

// Fragment loader: 16 bf16 per lane as two contiguous 16B loads, per ISA 7.12.2
// (lanes 0-15: K = {0..7,16..23}; lanes 16-31: K = {8..15,24..31}; caller adds half*8)
__device__ __forceinline__ v16bf load_frag(const __bf16* __restrict__ p) {
    v8bf lo = *(const v8bf*)p;
    v8bf hi = *(const v8bf*)(p + 16);
    return __builtin_shufflevector(lo, hi, 0, 1, 2, 3, 4, 5, 6, 7,
                                           8, 9, 10, 11, 12, 13, 14, 15);
}

// ---------------- WMMA GEMM: C[M,N](bf16) = A[M,K](bf16,row) @ Bt[N,K](bf16,row)^T
// grid = (M/16, N/256), block = 128 (4 waves). Each wave: a 16x64 strip with 4
// accumulators reusing one A fragment; K loop step 32 (v_wmma_f32_16x16x32_bf16),
// software-pipelined: next k-step's A + 4 B fragments are in flight while the
// current 4 WMMAs execute, so the loadcnt wait covers a full k-step of latency.
__global__ void __launch_bounds__(128)
k_gemm_bf16(const __bf16* __restrict__ A, const __bf16* __restrict__ Bt,
            __bf16* __restrict__ C, int N, int K) {
    const int lane = threadIdx.x & 31;
    const int wave = threadIdx.x >> 5;
    const int l16  = lane & 15;
    const int half = lane >> 4;
    const int rowM = blockIdx.x * 16;
    const int colBase = (blockIdx.y * 4 + wave) * 64;

    const __bf16* Ap  = A + (size_t)(rowM + l16) * K + half * 8;
    const __bf16* Bp0 = Bt + (size_t)(colBase + l16) * K + half * 8;
    const size_t  bstride = (size_t)16 * K;   // 16 output columns apart

    v8f acc0 = {0,0,0,0,0,0,0,0}, acc1 = {0,0,0,0,0,0,0,0};
    v8f acc2 = {0,0,0,0,0,0,0,0}, acc3 = {0,0,0,0,0,0,0,0};

    v16bf a_c  = load_frag(Ap);
    v16bf b_c0 = load_frag(Bp0);
    v16bf b_c1 = load_frag(Bp0 + bstride);
    v16bf b_c2 = load_frag(Bp0 + 2 * bstride);
    v16bf b_c3 = load_frag(Bp0 + 3 * bstride);

    for (int k0 = 32; k0 < K; k0 += 32) {
        v16bf a_n  = load_frag(Ap + k0);
        v16bf b_n0 = load_frag(Bp0 + k0);
        v16bf b_n1 = load_frag(Bp0 + bstride + k0);
        v16bf b_n2 = load_frag(Bp0 + 2 * bstride + k0);
        v16bf b_n3 = load_frag(Bp0 + 3 * bstride + k0);

        acc0 = __builtin_amdgcn_wmma_f32_16x16x32_bf16(false, a_c, false, b_c0, (short)0, acc0, false, false);
        acc1 = __builtin_amdgcn_wmma_f32_16x16x32_bf16(false, a_c, false, b_c1, (short)0, acc1, false, false);
        acc2 = __builtin_amdgcn_wmma_f32_16x16x32_bf16(false, a_c, false, b_c2, (short)0, acc2, false, false);
        acc3 = __builtin_amdgcn_wmma_f32_16x16x32_bf16(false, a_c, false, b_c3, (short)0, acc3, false, false);

        a_c = a_n; b_c0 = b_n0; b_c1 = b_n1; b_c2 = b_n2; b_c3 = b_n3;
    }
    // pipeline tail
    acc0 = __builtin_amdgcn_wmma_f32_16x16x32_bf16(false, a_c, false, b_c0, (short)0, acc0, false, false);
    acc1 = __builtin_amdgcn_wmma_f32_16x16x32_bf16(false, a_c, false, b_c1, (short)0, acc1, false, false);
    acc2 = __builtin_amdgcn_wmma_f32_16x16x32_bf16(false, a_c, false, b_c2, (short)0, acc2, false, false);
    acc3 = __builtin_amdgcn_wmma_f32_16x16x32_bf16(false, a_c, false, b_c3, (short)0, acc3, false, false);

    // C/D layout: VGPR r, lanes 0-15 -> M=r, lanes 16-31 -> M=r+8; N = lane&15
    v8f accs[4] = {acc0, acc1, acc2, acc3};
    #pragma unroll
    for (int t = 0; t < 4; ++t) {
        int col = colBase + t * 16 + l16;
        #pragma unroll
        for (int r = 0; r < 8; ++r) {
            int orow = rowM + r + half * 8;
            C[(size_t)orow * N + col] = (__bf16)accs[t][r];
        }
    }
}

// ---------------- attention score vectors ----------------
// layer 1: s_src1[h*NNODES+n] = z1[n, h*128+:128] . a1[h, 0:128]; s_dst uses a1[h,128:256]
__global__ void k_scores1(const __bf16* __restrict__ z1, const float* __restrict__ a1,
                          float* __restrict__ s_src, float* __restrict__ s_dst) {
    int i = blockIdx.x * blockDim.x + threadIdx.x;
    if (i >= NNODES * 8) return;
    int h = i / NNODES, n = i - h * NNODES;
    const __bf16* zp = z1 + (size_t)n * 1024 + h * 128;
    const float*  av = a1 + h * 256;
    float ss = 0.f, sd = 0.f;
    for (int d = 0; d < 128; ++d) {
        float z = (float)zp[d];
        ss += z * av[d];
        sd += z * av[128 + d];
    }
    s_src[i] = ss; s_dst[i] = sd;
}

// layer 2: D=256, a2 has 512 entries
__global__ void k_scores2(const __bf16* __restrict__ z2, const float* __restrict__ a2,
                          float* __restrict__ s_src, float* __restrict__ s_dst) {
    int n = blockIdx.x * blockDim.x + threadIdx.x;
    if (n >= NNODES) return;
    const __bf16* zp = z2 + (size_t)n * 256;
    float ss = 0.f, sd = 0.f;
    for (int d = 0; d < 256; ++d) {
        float z = (float)zp[d];
        ss += z * a2[d];
        sd += z * a2[256 + d];
    }
    s_src[n] = ss; s_dst[n] = sd;
}

// ---------------- edge softmax passes ----------------
__global__ void k_edge_max(const int* __restrict__ src, const int* __restrict__ dst,
                           const float* __restrict__ s_src, const float* __restrict__ s_dst,
                           unsigned* __restrict__ m_u, int E, int H) {
    int i = blockIdx.x * blockDim.x + threadIdx.x;
    if (i >= E * H) return;
    int e = i / H, h = i - e * H;
    int s = src[e], d = dst[e];
    float v = lrelu(s_src[h * NNODES + s] + s_dst[h * NNODES + d]);
    atomicMax(&m_u[h * NNODES + d], f2ord(v));
}

__global__ void k_edge_sum(const int* __restrict__ src, const int* __restrict__ dst,
                           const float* __restrict__ s_src, const float* __restrict__ s_dst,
                           const unsigned* __restrict__ m_u, float* __restrict__ den,
                           int E, int H) {
    int i = blockIdx.x * blockDim.x + threadIdx.x;
    if (i >= E * H) return;
    int e = i / H, h = i - e * H;
    int s = src[e], d = dst[e];
    float v = lrelu(s_src[h * NNODES + s] + s_dst[h * NNODES + d]);
    float m = ord2f(m_u[h * NNODES + d]);
    atomicAdd(&den[h * NNODES + d], __expf(v - m));
}

// one block per edge; blockDim = H*D (1024 for layer 1, 256 for layer 2)
// dshift = log2(D)
__global__ void k_edge_agg(const int* __restrict__ src, const int* __restrict__ dst,
                           const float* __restrict__ s_src, const float* __restrict__ s_dst,
                           const unsigned* __restrict__ m_u, const float* __restrict__ den,
                           const __bf16* __restrict__ z, float* __restrict__ hout,
                           int dshift, int H) {
    __shared__ float alpha_sh[8];
    __shared__ int ssh, dsh;
    int e = blockIdx.x;
    if (threadIdx.x == 0) { ssh = src[e]; dsh = dst[e]; }
    __syncthreads();
    int s = ssh, d = dsh;
    if ((int)threadIdx.x < H) {
        int h = threadIdx.x;
        float v = lrelu(s_src[h * NNODES + s] + s_dst[h * NNODES + d]);
        float m = ord2f(m_u[h * NNODES + d]);
        alpha_sh[h] = __expf(v - m) / den[h * NNODES + d];
    }
    __syncthreads();
    int t = threadIdx.x;
    int R = H << dshift;
    int h = t >> dshift;
    float zval = (float)z[(size_t)s * R + t];
    atomicAdd(&hout[(size_t)d * R + t], alpha_sh[h] * zval);
}

// h1 = bf16(leaky_relu(hout1))
__global__ void k_lrelu_bf16(const float* __restrict__ in, __bf16* __restrict__ out, int n) {
    int i = blockIdx.x * blockDim.x + threadIdx.x;
    if (i < n) out[i] = (__bf16)lrelu(in[i]);
}

// ---------------- final edge scoring ----------------
// score[e] = h2[src[e]] @ Wp[:256] + h2[dst[e]] @ Wp[256:] + bp   (2 outputs)
__global__ void __launch_bounds__(256)
k_score(const float* __restrict__ h2, const float* __restrict__ Wp,
        const float* __restrict__ bp, const int* __restrict__ src,
        const int* __restrict__ dst, float* __restrict__ out, int E) {
    __shared__ float wp_sh[1024];  // Wp is (512, 2) row-major
    for (int j = threadIdx.x; j < 1024; j += 256) wp_sh[j] = Wp[j];
    __syncthreads();
    int e = blockIdx.x * blockDim.x + threadIdx.x;
    if (e >= E) return;
    int s = src[e], d = dst[e];
    const float* hs = h2 + (size_t)s * 256;
    const float* hd = h2 + (size_t)d * 256;
    float o0 = bp[0], o1 = bp[1];
    for (int k = 0; k < 256; ++k) {
        float x = hs[k], y = hd[k];
        o0 += x * wp_sh[k * 2 + 0] + y * wp_sh[(256 + k) * 2 + 0];
        o1 += x * wp_sh[k * 2 + 1] + y * wp_sh[(256 + k) * 2 + 1];
    }
    out[(size_t)e * 2 + 0] = o0;
    out[(size_t)e * 2 + 1] = o1;
}

// ---------------- host orchestration ----------------
extern "C" void kernel_launch(void* const* d_in, const int* in_sizes, int n_in,
                              void* d_out, int out_size, void* d_ws, size_t ws_size,
                              hipStream_t stream) {
    (void)n_in; (void)out_size; (void)ws_size;
    const float* h_in = (const float*)d_in[0];
    const float* W1   = (const float*)d_in[1];
    const float* a1   = (const float*)d_in[2];
    const float* W2   = (const float*)d_in[3];
    const float* a2   = (const float*)d_in[4];
    const float* Wp   = (const float*)d_in[5];
    const float* bp   = (const float*)d_in[6];
    const int* src_mp = (const int*)d_in[7];
    const int* dst_mp = (const int*)d_in[8];
    const int* src_f  = (const int*)d_in[9];
    const int* dst_f  = (const int*)d_in[10];
    const int E_mp   = in_sizes[7];
    const int E_full = in_sizes[9];

    char* ws = (char*)d_ws;
    size_t off = 0;
    auto carve = [&](size_t bytes) -> char* {
        char* p = ws + off;
        off = (off + bytes + 255) & ~(size_t)255;
        return p;
    };
    __bf16*   h_bf   = (__bf16*)carve((size_t)NNODES * 256 * 2);   // 10.24 MB
    __bf16*   Wt1    = (__bf16*)carve((size_t)1024 * 256 * 2);     // 0.52 MB
    __bf16*   Wt2    = (__bf16*)carve((size_t)256 * 1024 * 2);     // 0.52 MB
    __bf16*   z1bf   = (__bf16*)carve((size_t)NNODES * 1024 * 2);  // 40.96 MB (later aliased as h1 bf16)
    float*    s_src1 = (float*)carve((size_t)8 * NNODES * 4);
    float*    s_dst1 = (float*)carve((size_t)8 * NNODES * 4);
    unsigned* m1     = (unsigned*)carve((size_t)8 * NNODES * 4);
    float*    den1   = (float*)carve((size_t)8 * NNODES * 4);
    float*    hout1  = (float*)carve((size_t)NNODES * 1024 * 4);   // 81.92 MB
    __bf16*   z2bf   = (__bf16*)carve((size_t)NNODES * 256 * 2);   // 10.24 MB
    float*    s_src2 = (float*)carve((size_t)NNODES * 4);
    float*    s_dst2 = (float*)carve((size_t)NNODES * 4);
    unsigned* m2     = (unsigned*)carve((size_t)NNODES * 4);
    float*    den2   = (float*)carve((size_t)NNODES * 4);
    float*    hout2  = (float*)carve((size_t)NNODES * 256 * 4);    // 20.48 MB
    __bf16*   h1bf   = z1bf;  // alias: z1 is dead once layer-1 aggregation finishes

    // zero accumulators (0u is also the identity of the ordered-max encoding)
    hipMemsetAsync(m1,    0, (size_t)8 * NNODES * 4, stream);
    hipMemsetAsync(den1,  0, (size_t)8 * NNODES * 4, stream);
    hipMemsetAsync(hout1, 0, (size_t)NNODES * 1024 * 4, stream);
    hipMemsetAsync(m2,    0, (size_t)NNODES * 4, stream);
    hipMemsetAsync(den2,  0, (size_t)NNODES * 4, stream);
    hipMemsetAsync(hout2, 0, (size_t)NNODES * 256 * 4, stream);

    // --- prep ---
    {
        int n = NNODES * 256;
        k_cvt_bf16<<<(n + 255) / 256, 256, 0, stream>>>(h_in, h_bf, n);
        k_prep_w1<<<(1024 * 256 + 255) / 256, 256, 0, stream>>>(W1, Wt1);
        k_prep_w2<<<(256 * 1024 + 255) / 256, 256, 0, stream>>>(W2, Wt2);
    }

    // --- layer 1: z1 = h @ [W1 heads]  (20000x256 @ 256x1024) ---
    k_gemm_bf16<<<dim3(NNODES / 16, 1024 / 256), 128, 0, stream>>>(h_bf, Wt1, z1bf, 1024, 256);
    k_scores1<<<(NNODES * 8 + 255) / 256, 256, 0, stream>>>(z1bf, a1, s_src1, s_dst1);
    k_edge_max<<<((size_t)E_mp * 8 + 255) / 256, 256, 0, stream>>>(src_mp, dst_mp, s_src1, s_dst1, m1, E_mp, 8);
    k_edge_sum<<<((size_t)E_mp * 8 + 255) / 256, 256, 0, stream>>>(src_mp, dst_mp, s_src1, s_dst1, m1, den1, E_mp, 8);
    k_edge_agg<<<E_mp, 1024, 0, stream>>>(src_mp, dst_mp, s_src1, s_dst1, m1, den1, z1bf, hout1, 7, 8);

    // h1 = bf16(leaky_relu(hout1)) -> reuse z1's buffer
    k_lrelu_bf16<<<(NNODES * 1024 + 255) / 256, 256, 0, stream>>>(hout1, h1bf, NNODES * 1024);

    // --- layer 2: z2 = h1 @ W2  (20000x1024 @ 1024x256) ---
    k_gemm_bf16<<<dim3(NNODES / 16, 256 / 256), 128, 0, stream>>>(h1bf, Wt2, z2bf, 256, 1024);
    k_scores2<<<(NNODES + 255) / 256, 256, 0, stream>>>(z2bf, a2, s_src2, s_dst2);
    k_edge_max<<<((size_t)E_mp + 255) / 256, 256, 0, stream>>>(src_mp, dst_mp, s_src2, s_dst2, m2, E_mp, 1);
    k_edge_sum<<<((size_t)E_mp + 255) / 256, 256, 0, stream>>>(src_mp, dst_mp, s_src2, s_dst2, m2, den2, E_mp, 1);
    k_edge_agg<<<E_mp, 256, 0, stream>>>(src_mp, dst_mp, s_src2, s_dst2, m2, den2, z2bf, hout2, 8, 1);

    // --- final edge scoring ---
    k_score<<<(E_full + 255) / 256, 256, 0, stream>>>(hout2, Wp, bp, src_f, dst_f, (float*)d_out, E_full);
}
